// MultiHeadAttention_5136780886321
// MI455X (gfx1250) — matrix operand order repaired and checked by reading, not compiled, not run on hardware
//
#include <hip/hip_runtime.h>
#include <cstdint>

#define DM  512   // d_model
#define NH  8     // heads
#define DH  64    // head dim
#define NB  4     // batch
#define SEQ 2048  // sequence length

typedef __attribute__((ext_vector_type(16))) _Float16 v16h;
typedef __attribute__((ext_vector_type(8)))  _Float16 v8h;
typedef __attribute__((ext_vector_type(8)))  float    v8f;
typedef __attribute__((ext_vector_type(4)))  unsigned int v4u;
typedef __attribute__((ext_vector_type(8)))  int      v8i;
typedef __attribute__((ext_vector_type(4)))  int      v4i;

// ---------------------------------------------------------------- WMMA helper
__device__ __forceinline__ v8f wmma_f16(v16h a, v16h b, v8f c) {
  // D = A(16x32 f16) * B(32x16 f16) + C(16x16 f32)
  return __builtin_amdgcn_wmma_f32_16x16x32_f16(false, a, false, b, (short)0, c,
                                                false, false);
}

__device__ __forceinline__ v8f v8f_zero() {
  v8f z = {0.f, 0.f, 0.f, 0.f, 0.f, 0.f, 0.f, 0.f};
  return z;
}

// A-fragment 16x32 f16: lane holds row = lane&15,
// halves 0..15 = K kb..kb+15 (contiguous at p).
__device__ __forceinline__ v16h frag_gap(const _Float16* p) {
  v8h lo = *(const v8h*)p;
  v8h hi = *(const v8h*)(p + 8);
  v16h a;
#pragma unroll
  for (int e = 0; e < 8; ++e) { a[e] = lo[e]; a[e + 8] = hi[e]; }
  return a;
}

// B-fragment 32x16 f16: lane holds col = lane&15, K = kb16 + (0..15),
// 16 contiguous halves starting at p.
__device__ __forceinline__ v16h frag_contig(const _Float16* p) {
  v8h lo = *(const v8h*)p;
  v8h hi = *(const v8h*)(p + 8);
  v16h b;
#pragma unroll
  for (int e = 0; e < 8; ++e) { b[e] = lo[e]; b[e + 8] = hi[e]; }
  return b;
}

// ---------------- VALU cross-lane butterflies via v_permlane16_b32 -----------
// Selectors encode dst-lane i <- src-lane (i ^ m) within each 16-lane row, so
// the reductions stay on the VALU pipe instead of ds_bpermute (LDS pipe).
__device__ __forceinline__ float permxor16(float v, int s0, int s1) {
  int x = __float_as_int(v);
  x = __builtin_amdgcn_permlane16(x, x, s0, s1, false, false);
  return __int_as_float(x);
}
__device__ __forceinline__ float red_max16(float v) {
  v = fmaxf(v, permxor16(v, (int)0x67452301, (int)0xEFCDAB89));  // xor 1
  v = fmaxf(v, permxor16(v, (int)0x54761032, (int)0xDCFE98BA));  // xor 2
  v = fmaxf(v, permxor16(v, (int)0x32107654, (int)0xBA98FEDC));  // xor 4
  v = fmaxf(v, permxor16(v, (int)0xFEDCBA98, (int)0x76543210));  // xor 8
  return v;
}
__device__ __forceinline__ float red_sum16(float v) {
  v += permxor16(v, (int)0x67452301, (int)0xEFCDAB89);
  v += permxor16(v, (int)0x54761032, (int)0xDCFE98BA);
  v += permxor16(v, (int)0x32107654, (int)0xBA98FEDC);
  v += permxor16(v, (int)0xFEDCBA98, (int)0x76543210);
  return v;
}

// ------------------------------------------------- Tensor Data Mover (2D tile)
// Loads a [rows x cols] tile of 2-byte elements (row stride = stride_elems)
// from global memory into LDS at lds_off, tracked by TENSORcnt.
__device__ __forceinline__ void tdm_load_2d(const void* gsrc, unsigned lds_off,
                                            unsigned rows, unsigned cols,
                                            unsigned stride_elems) {
  unsigned long long ga = (unsigned long long)(uintptr_t)gsrc;
  v4u g0;
  g0.x = 1u;                                                 // count=1 (user D#)
  g0.y = lds_off;                                            // lds_addr (bytes)
  g0.z = (unsigned)ga;                                       // global_addr[31:0]
  g0.w = (unsigned)((ga >> 32) & 0x01FFFFFFu) | (2u << 30);  // addr[56:32]|type=2
  v8i g1;
  g1[0] = (int)(1u << 16);                    // wg_mask=0 | data_size=1 (2 bytes)
  g1[1] = (int)(cols << 16);                  // tensor_dim0[15:0]
  g1[2] = (int)((cols >> 16) | (rows << 16)); // tensor_dim0[31:16]|tensor_dim1 lo
  g1[3] = (int)((rows >> 16) | (cols << 16)); // tensor_dim1 hi | tile_dim0
  g1[4] = (int)(rows & 0xFFFFu);              // tile_dim1 | tile_dim2=0
  g1[5] = (int)stride_elems;                  // tensor_dim0_stride[31:0]
  g1[6] = 0;                                  // stride0[47:32] | dim1_stride lo
  g1[7] = 0;
  v4i z4 = {0, 0, 0, 0};
#if defined(__clang_major__) && (__clang_major__ >= 23)
  v8i z8 = {0, 0, 0, 0, 0, 0, 0, 0};
  __builtin_amdgcn_tensor_load_to_lds(g0, g1, z4, z4, z8, 0);
#else
  __builtin_amdgcn_tensor_load_to_lds(g0, g1, z4, z4, 0);
#endif
}

// --------------------------------------------------------------- GEMM + bias
// Y(Mr x 512) = X(Mr x 512) * W(512 x 512) + bias ; all WMMA f16 with f32 acc.
// VTRANS: store V projection transposed as vT[((b*NH+h)*DH+d)*SEQ + m] (f16)
// so the attention kernel can TDM-load d-major V tiles.
template <typename TIN, typename TOUT, bool VTRANS>
__global__ void __launch_bounds__(256)
gemm_bias_kernel(const TIN* __restrict__ X, const float* __restrict__ W,
                 const float* __restrict__ bias, TOUT* __restrict__ Y) {
  __shared__ _Float16 Xs[64 * 32];     // [row][k]
  __shared__ _Float16 Wst[128 * 32];   // transposed: [n][k]

  const int tid  = threadIdx.x;
  const int lane = tid & 31;
  const int wave = tid >> 5;
  const int l15  = lane & 15;
  const int hi16 = (lane >> 4) & 1;
  const int kb8  = hi16 ? 8 : 0;
  const int kb16 = hi16 ? 16 : 0;
  const int row0 = blockIdx.x * 64;
  const int col0 = blockIdx.y * 128;
  const int wr   = (wave >> 2) * 32;   // 2 wave-rows
  const int wc   = (wave & 3) * 32;    // 4 wave-cols

  v8f c00 = v8f_zero(), c01 = v8f_zero(), c10 = v8f_zero(), c11 = v8f_zero();

  for (int kc = 0; kc < DM; kc += 32) {
    __syncthreads();
    {  // stage X tile (f16) and W tile (transposed f16)
      int xr = tid >> 2, xc = (tid & 3) * 8;
      const TIN* src = X + (size_t)(row0 + xr) * DM + kc + xc;
#pragma unroll
      for (int i = 0; i < 8; ++i) Xs[xr * 32 + xc + i] = (_Float16)src[i];
      int wk = tid >> 3, wn = (tid & 7) * 16;
      const float* wsrc = W + (size_t)(kc + wk) * DM + col0 + wn;
#pragma unroll
      for (int i = 0; i < 16; ++i) Wst[(wn + i) * 32 + wk] = (_Float16)wsrc[i];
      if (kc + 32 < DM) {  // speculative prefetch of next k-chunk
        __builtin_prefetch(src + 32, 0, 1);
        __builtin_prefetch(wsrc + 32 * DM, 0, 1);
      }
    }
    __syncthreads();

    v16h a0 = frag_gap(&Xs[(wr + l15) * 32 + kb16]);
    v16h a1 = frag_gap(&Xs[(wr + 16 + l15) * 32 + kb16]);
    v16h b0 = frag_contig(&Wst[(wc + l15) * 32 + kb16]);
    v16h b1 = frag_contig(&Wst[(wc + 16 + l15) * 32 + kb16]);
    c00 = wmma_f16(a0, b0, c00);
    c01 = wmma_f16(a0, b1, c01);
    c10 = wmma_f16(a1, b0, c10);
    c11 = wmma_f16(a1, b1, c11);
  }

  // C layout: reg r, lanes0-15 -> row r, lanes16-31 -> row 8+r; col = lane&15.
  auto store1 = [&](int gr, int gc, float v) {
    if constexpr (VTRANS) {
      int bb = gr >> 11, mm = gr & (SEQ - 1);
      int hh = gc >> 6,  dd = gc & (DH - 1);
      Y[((size_t)((bb * NH + hh) * DH + dd)) * SEQ + mm] = (TOUT)v;
    } else {
      Y[(size_t)gr * DM + gc] = (TOUT)v;
    }
  };
#pragma unroll
  for (int r = 0; r < 8; ++r) {
    int gr0 = row0 + wr + hi16 * 8 + r;
    int gc0 = col0 + wc + l15;
    store1(gr0,      gc0,      c00[r] + bias[gc0]);
    store1(gr0,      gc0 + 16, c01[r] + bias[gc0 + 16]);
    store1(gr0 + 16, gc0,      c10[r] + bias[gc0]);
    store1(gr0 + 16, gc0 + 16, c11[r] + bias[gc0 + 16]);
  }
}

// ------------------------------------------------------------ Flash attention
// Grid: (SEQ/256, NH, NB), 256 threads = 8 waves, wave owns 32 q rows
// (two 16-row tiles processed sequentially to bound register pressure).
// K/V kv-tiles (32 rows) double-buffered in LDS via the Tensor Data Mover.
__global__ void __launch_bounds__(256)
attn_kernel(const _Float16* __restrict__ qh, const _Float16* __restrict__ kh,
            const _Float16* __restrict__ vT, _Float16* __restrict__ oh) {
  __shared__ _Float16 Kt[2][32 * 64];   // [kv][d]
  __shared__ _Float16 Vt[2][64 * 32];   // [d][kv]  (pre-transposed in ws)
  __shared__ _Float16 Ps[16][16 * 32];  // per (wave, row-tile) P staging
  __shared__ _Float16 Qs[8][32 * 64];   // per-wave Q tile (32 x 64)

  const int tid  = threadIdx.x;
  const int lane = tid & 31;
  const int wave = tid >> 5;
  const int l15  = lane & 15;
  const int hi16 = (lane >> 4) & 1;
  const int kb8  = hi16 ? 8 : 0;
  const int kb16 = hi16 ? 16 : 0;
  const int b = blockIdx.z, h = blockIdx.y;
  const int q0 = blockIdx.x * 256 + wave * 32;

  // Stage this wave's Q tile (32 x 64 halves): lane copies one 128B row.
  {
    const _Float16* src = qh + (size_t)(b * SEQ + q0 + lane) * DM + h * DH;
    const v8h* s8 = (const v8h*)src;
    v8h* d8 = (v8h*)&Qs[wave][lane * 64];
#pragma unroll
    for (int i = 0; i < 8; ++i) d8[i] = s8[i];
  }
  // A-fragments with the softmax scale (1/sqrt(M), per reference) pre-folded.
  const _Float16 hscale = (_Float16)0.02209708691f;
  v16h qa[2][2];
#pragma unroll
  for (int t = 0; t < 2; ++t)
#pragma unroll
    for (int c = 0; c < 2; ++c) {
      v16h a = frag_gap(&Qs[wave][(t * 16 + l15) * 64 + c * 32 + kb16]);
#pragma unroll
      for (int e = 0; e < 16; ++e) a[e] = a[e] * hscale;
      qa[t][c] = a;
    }

  const _Float16* kbase = kh + (size_t)(b * SEQ) * DM + h * DH;
  const _Float16* vbase = vT + (size_t)((b * NH + h) * DH) * SEQ;

  int buf = 0;
  if (wave == 0) {  // prologue: TDM-load kv-block 0
    tdm_load_2d(kbase, (unsigned)(uintptr_t)(void*)&Kt[0][0], 32, 64, DM);
    tdm_load_2d(vbase, (unsigned)(uintptr_t)(void*)&Vt[0][0], 64, 32, SEQ);
  }

  v8f o[2][4];
  float mrow[2][8], lrow[2][8];
#pragma unroll
  for (int t = 0; t < 2; ++t)
#pragma unroll
    for (int r = 0; r < 8; ++r) {
      mrow[t][r] = -1e30f; lrow[t][r] = 0.f;
      if (r < 4) o[t][r] = v8f_zero();
    }

  const int NKV = SEQ / 32;
  for (int it = 0; it < NKV; ++it) {
    if (wave == 0) __builtin_amdgcn_s_wait_tensorcnt(0);
    __syncthreads();  // tile `it` now valid in buf for every wave
    if (wave == 0 && it + 1 < NKV) {  // prefetch next kv-block into other buf
      tdm_load_2d(kbase + (size_t)(it + 1) * 32 * DM,
                  (unsigned)(uintptr_t)(void*)&Kt[buf ^ 1][0], 32, 64, DM);
      tdm_load_2d(vbase + (it + 1) * 32,
                  (unsigned)(uintptr_t)(void*)&Vt[buf ^ 1][0], 64, 32, SEQ);
    }
    const _Float16* Ktb = &Kt[buf][0];
    const _Float16* Vtb = &Vt[buf][0];

#pragma unroll
    for (int t = 0; t < 2; ++t) {
      // S(16x32) = (scale*Q)(16x64) * K^T ; B[d][kv]=Kt[kv][d], contiguous in d.
      v8f s0 = v8f_zero(), s1 = v8f_zero();
      s0 = wmma_f16(qa[t][0], frag_contig(Ktb + (l15) * 64 +  0 + kb16), s0);
      s0 = wmma_f16(qa[t][1], frag_contig(Ktb + (l15) * 64 + 32 + kb16), s0);
      s1 = wmma_f16(qa[t][0], frag_contig(Ktb + (16 + l15) * 64 +  0 + kb16), s1);
      s1 = wmma_f16(qa[t][1], frag_contig(Ktb + (16 + l15) * 64 + 32 + kb16), s1);

      // Online softmax (rows split across lane halves; VALU butterflies).
      _Float16* Pw = &Ps[wave * 2 + t][0];
#pragma unroll
      for (int r = 0; r < 8; ++r) {
        float a0 = s0[r], a1 = s1[r];
        float mn = fmaxf(mrow[t][r], red_max16(fmaxf(a0, a1)));
        float alpha = __expf(mrow[t][r] - mn);
        float p0 = __expf(a0 - mn), p1 = __expf(a1 - mn);
        lrow[t][r] = lrow[t][r] * alpha + red_sum16(p0 + p1);
        mrow[t][r] = mn;
        o[t][0][r] *= alpha; o[t][1][r] *= alpha;
        o[t][2][r] *= alpha; o[t][3][r] *= alpha;
        int row = r + hi16 * 8;
        Pw[row * 32 + l15]      = (_Float16)p0;
        Pw[row * 32 + 16 + l15] = (_Float16)p1;
      }

      // O(16x64) += P(16x32)*V(32x64); B[kv][d]=Vt[d][kv], contiguous in kv.
      v16h pa = frag_gap(&Pw[l15 * 32 + kb16]);
      o[t][0] = wmma_f16(pa, frag_contig(Vtb + (     l15) * 32 + kb16), o[t][0]);
      o[t][1] = wmma_f16(pa, frag_contig(Vtb + (16 + l15) * 32 + kb16), o[t][1]);
      o[t][2] = wmma_f16(pa, frag_contig(Vtb + (32 + l15) * 32 + kb16), o[t][2]);
      o[t][3] = wmma_f16(pa, frag_contig(Vtb + (48 + l15) * 32 + kb16), o[t][3]);
    }

    __syncthreads();  // done reading buf before TDM (it+2) may overwrite it
    buf ^= 1;
  }

  // Normalize and write O in (b, m, h, d) layout (== reshape(b,m,H*DV)).
#pragma unroll
  for (int t = 0; t < 2; ++t)
#pragma unroll
    for (int r = 0; r < 8; ++r) {
      float inv = 1.0f / lrow[t][r];
      size_t base =
          (size_t)(b * SEQ + q0 + t * 16 + hi16 * 8 + r) * DM + h * DH;
      oh[base +      l15] = (_Float16)(o[t][0][r] * inv);
      oh[base + 16 + l15] = (_Float16)(o[t][1][r] * inv);
      oh[base + 32 + l15] = (_Float16)(o[t][2][r] * inv);
      oh[base + 48 + l15] = (_Float16)(o[t][3][r] * inv);
    }
}

// ------------------------------------------------------------------- launcher
extern "C" void kernel_launch(void* const* d_in, const int* in_sizes, int n_in,
                              void* d_out, int out_size, void* d_ws,
                              size_t ws_size, hipStream_t stream) {
  const float* query = (const float*)d_in[0];
  const float* key_  = (const float*)d_in[1];
  const float* value = (const float*)d_in[2];
  const float* Wq = (const float*)d_in[3];
  const float* bq = (const float*)d_in[4];
  const float* Wk = (const float*)d_in[5];
  const float* bk = (const float*)d_in[6];
  const float* Wv = (const float*)d_in[7];
  const float* bv = (const float*)d_in[8];
  const float* Wo = (const float*)d_in[9];
  const float* bo = (const float*)d_in[10];
  float* out = (float*)d_out;

  char* ws = (char*)d_ws;
  const size_t SLICE = (size_t)NB * SEQ * DM * sizeof(_Float16);  // 8 MiB
  _Float16* qh = (_Float16*)(ws);
  _Float16* kh = (_Float16*)(ws + SLICE);
  _Float16* vT = (_Float16*)(ws + 2 * SLICE);
  _Float16* oh = (_Float16*)(ws + 3 * SLICE);

  const int Mr = NB * SEQ;  // 8192 rows
  dim3 blk(256);
  dim3 ggrid(Mr / 64, DM / 128);

  gemm_bias_kernel<float, _Float16, false>
      <<<ggrid, blk, 0, stream>>>(query, Wq, bq, qh);
  gemm_bias_kernel<float, _Float16, false>
      <<<ggrid, blk, 0, stream>>>(key_, Wk, bk, kh);
  gemm_bias_kernel<float, _Float16, true>
      <<<ggrid, blk, 0, stream>>>(value, Wv, bv, vT);

  attn_kernel<<<dim3(SEQ / 256, NH, NB), blk, 0, stream>>>(qh, kh, vT, oh);

  gemm_bias_kernel<_Float16, float, false>
      <<<ggrid, blk, 0, stream>>>(oh, Wo, bo, out);
}
